// GAT_4879082848783
// MI455X (gfx1250) — matrix operand order repaired and checked by reading, not compiled, not run on hardware
//
#include <hip/hip_runtime.h>
#include <hip/hip_bf16.h>

typedef __attribute__((ext_vector_type(16))) _Float16 v16h;
typedef __attribute__((ext_vector_type(8)))  float    v8f;

#define GAT_B 8
#define GAT_N 2048
#define GAT_D 64
#define NEG_SLOPE 0.2f
#define NEG_INF_F (-9.0e15f)

// ---------------------------------------------------------------------------
// Kernel 1: Wh = X @ Ws  (f16 WMMA, f32 accumulate)
// One wave per 16-row tile. Stores Wh (f32, row major [16384,64]) and
// WhT (f16, [B][64][2048]) for fast B-matrix tile loads in the attention pass.
// ---------------------------------------------------------------------------
__global__ __launch_bounds__(128) void wh_kernel(const float* __restrict__ X,
                                                 const float* __restrict__ Ws,
                                                 float* __restrict__ Wh,
                                                 _Float16* __restrict__ WhT) {
    const int lane = threadIdx.x & 31;
    const int wave = threadIdx.x >> 5;
    const int tile = blockIdx.x * 4 + wave;          // 0..1023
    const int i0   = tile << 4;                      // global row base
    const int m    = lane & 15;
    const int half = lane >> 4;
    const int n    = m;                              // column lane for B/C tiles

    // A-matrix tiles: X[i0..i0+15][0..63] as two 16x32 f16 tiles (ISA layout)
    v16h xa[2];
#pragma unroll
    for (int kc = 0; kc < 2; ++kc) {
#pragma unroll
        for (int j = 0; j < 8; ++j) {
            const int k = kc * 32 + ((j >> 2) << 4) + (half << 3) + ((j & 3) << 1);
            const float2 xv = *(const float2*)(X + (size_t)(i0 + m) * 64 + k);
            xa[kc][2 * j]     = (_Float16)xv.x;
            xa[kc][2 * j + 1] = (_Float16)xv.y;
        }
    }

    v8f acc[4] = {};
#pragma unroll
    for (int t = 0; t < 4; ++t) {
#pragma unroll
        for (int kc = 0; kc < 2; ++kc) {
            // B-matrix 32x16 tile of Ws (row major): VGPR r holds rows
            // K = (r/4)*16 + half*8 + 2(r%4), +1, column = n
            v16h bt;
#pragma unroll
            for (int r = 0; r < 8; ++r) {
                const int kk = kc * 32 + ((r >> 2) << 4) + (half << 3) + ((r & 3) << 1);
                bt[2 * r]     = (_Float16)Ws[(size_t)kk * 64 + t * 16 + n];
                bt[2 * r + 1] = (_Float16)Ws[(size_t)(kk + 1) * 64 + t * 16 + n];
            }
            acc[t] = __builtin_amdgcn_wmma_f32_16x16x32_f16(
                false, xa[kc], false, bt, (short)0, acc[t], false, false);
        }
    }

    // C/D layout: VGPR r -> row (r + 8*half), column n
#pragma unroll
    for (int r = 0; r < 8; ++r) {
        const int g    = i0 + r + (half << 3);       // global row
        const int bb   = g >> 11;                    // batch
        const int jloc = g & 2047;                   // row within batch
#pragma unroll
        for (int t = 0; t < 4; ++t) {
            const float v = acc[t][r];
            Wh[(size_t)g * 64 + t * 16 + n] = v;
            WhT[((size_t)bb * 64 + t * 16 + n) * 2048 + jloc] = (_Float16)v;
        }
    }
}

// ---------------------------------------------------------------------------
// Kernel 2: Wh1 = Wh . a[:64], Wh2 = Wh . a[64:]
// ---------------------------------------------------------------------------
__global__ __launch_bounds__(256) void wh12_kernel(const float* __restrict__ Wh,
                                                   const float* __restrict__ av,
                                                   float* __restrict__ Wh1,
                                                   float* __restrict__ Wh2) {
    const int g = blockIdx.x * blockDim.x + threadIdx.x;   // 0..16383
    const float* row = Wh + (size_t)g * 64;
    float s1 = 0.f, s2 = 0.f;
#pragma unroll 8
    for (int c = 0; c < 64; ++c) {
        const float v = row[c];
        s1 += v * av[c];
        s2 += v * av[64 + c];
    }
    Wh1[g] = s1;
    Wh2[g] = s2;
}

// ---------------------------------------------------------------------------
// Kernel 3: fused masked-softmax attention + aggregation (flash style),
// j-dimension split across the 4 waves of a block for 4x wave parallelism
// (4096 waves total -> keeps the 23.3 TB/s stream of A latency-covered).
// Per-wave partials (m, l, acc) merged through LDS (flash-decoding merge).
// ---------------------------------------------------------------------------
__global__ __launch_bounds__(128) void gat_kernel(const float* __restrict__ A,
                                                  const float* __restrict__ Wh1,
                                                  const float* __restrict__ Wh2,
                                                  const _Float16* __restrict__ WhT,
                                                  float* __restrict__ out) {
    __shared__ float ls_acc[4][16][64];   // 16 KB
    __shared__ float ls_m[4][16];
    __shared__ float ls_l[4][16];

    const int lane  = threadIdx.x & 31;
    const int wave  = threadIdx.x >> 5;
    const int tile  = blockIdx.x;                    // 0..1023 (one 16-row tile)
    const int b     = tile >> 7;                     // 128 tiles per batch
    const int i0loc = (tile & 127) << 4;
    const int m     = lane & 15;
    const int half  = lane >> 4;
    const int n     = m;

    const int g0 = b * GAT_N + i0loc;
    const float* Arow  = A + (size_t)(g0 + m) * GAT_N;   // this lane's query row of A
    const float* wh2b  = Wh2 + b * GAT_N;
    const float  wh1m  = Wh1[g0 + m];
    const _Float16* wp = WhT + ((size_t)b * 64 + n) * 2048 + (half << 4);

    const int j0 = wave << 9;                        // 512-column segment per wave

    float rm = -3.0e38f;     // running row max (this segment)
    float rl = 0.0f;         // running row sum (this segment)
    v8f acc[4] = {};

    for (int jbase = j0; jbase < j0 + 512; jbase += 32) {
        // hint-prefetch 2 chunks ahead of this lane's A row (stream-once data)
        if (jbase + 64 < GAT_N)
            __builtin_prefetch(Arow + jbase + 64, 0, 0);

        // --- scores in the f16 A-matrix slot order (lane = row m) ---
        float s[16];
#pragma unroll
        for (int j = 0; j < 8; ++j) {
            const int kidx = (half << 4) + (j << 1);
            const int jj   = jbase + kidx;
            const float2 av = *(const float2*)(Arow + jj);
            float e0 = wh1m + wh2b[jj];
            float e1 = wh1m + wh2b[jj + 1];
            e0 = fmaxf(e0, NEG_SLOPE * e0);          // leaky relu (slope<1)
            e1 = fmaxf(e1, NEG_SLOPE * e1);
            s[2 * j]     = (av.x > 0.f) ? av.x * e0 : NEG_INF_F;
            s[2 * j + 1] = (av.y > 0.f) ? av.y * e1 : NEG_INF_F;
        }

        // --- online softmax statistics (rows split across lane and lane^16) ---
        float cmax = s[0];
#pragma unroll
        for (int k = 1; k < 16; ++k) cmax = fmaxf(cmax, s[k]);
        cmax = fmaxf(cmax, __shfl_xor(cmax, 16));
        const float mnew  = fmaxf(rm, cmax);
        const float scale = __expf(rm - mnew);

        float p[16];
        float psum = 0.f;
#pragma unroll
        for (int k = 0; k < 16; ++k) {
            p[k] = __expf(s[k] - mnew);
            psum += p[k];
        }
        psum += __shfl_xor(psum, 16);
        rl = rl * scale + psum;
        rm = mnew;

        // rescale accumulators: C-layout VGPR r holds row (r + 8*half)
#pragma unroll
        for (int r = 0; r < 8; ++r) {
            const float al = __shfl(scale, r + (half << 3));
            acc[0][r] *= al; acc[1][r] *= al; acc[2][r] *= al; acc[3][r] *= al;
        }

        // pack P as the 16x32 f16 A-matrix (already in slot order)
        v16h pa;
#pragma unroll
        for (int k = 0; k < 16; ++k) pa[k] = (_Float16)p[k];

        // B tiles: WhT[b][c][jbase + half*16 + 0..15] -> 32B contiguous load
#pragma unroll
        for (int t = 0; t < 4; ++t) {
            const v16h bt = *(const v16h*)(wp + (size_t)t * 16 * 2048 + jbase);
            acc[t] = __builtin_amdgcn_wmma_f32_16x16x32_f16(
                false, pa, false, bt, (short)0, acc[t], false, false);
        }
    }

    // --- publish per-wave partials to LDS ---
    if (half == 0) {
        ls_m[wave][m] = rm;
        ls_l[wave][m] = rl;
    }
#pragma unroll
    for (int r = 0; r < 8; ++r) {
#pragma unroll
        for (int t = 0; t < 4; ++t)
            ls_acc[wave][r + (half << 3)][t * 16 + n] = acc[t][r];
    }
    __syncthreads();

    // --- merge the 4 segment-partials (flash-decoding combine) + ELU + store ---
    const int c  = threadIdx.x & 63;                 // output column
    const int rg = threadIdx.x >> 6;                 // row group (0 or 1)
#pragma unroll
    for (int rr = 0; rr < 8; ++rr) {
        const int rho = rg * 8 + rr;
        const float m0 = ls_m[0][rho], m1 = ls_m[1][rho];
        const float m2 = ls_m[2][rho], m3 = ls_m[3][rho];
        const float M  = fmaxf(fmaxf(m0, m1), fmaxf(m2, m3));
        const float f0 = __expf(m0 - M), f1 = __expf(m1 - M);
        const float f2 = __expf(m2 - M), f3 = __expf(m3 - M);
        const float l  = f0 * ls_l[0][rho] + f1 * ls_l[1][rho]
                       + f2 * ls_l[2][rho] + f3 * ls_l[3][rho];
        float v = f0 * ls_acc[0][rho][c] + f1 * ls_acc[1][rho][c]
                + f2 * ls_acc[2][rho][c] + f3 * ls_acc[3][rho][c];
        v = v / l;
        v = (v > 0.f) ? v : (__expf(v) - 1.0f);      // elu, alpha=1
        out[(size_t)(g0 + rho) * 64 + c] = v;
    }
}

// ---------------------------------------------------------------------------
extern "C" void kernel_launch(void* const* d_in, const int* in_sizes, int n_in,
                              void* d_out, int out_size, void* d_ws, size_t ws_size,
                              hipStream_t stream) {
    (void)in_sizes; (void)n_in; (void)out_size; (void)ws_size;
    const float* A  = (const float*)d_in[0];   // [8,2048,2048]
    const float* X  = (const float*)d_in[1];   // [8,2048,64]
    const float* Ws = (const float*)d_in[2];   // [64,64]
    const float* av = (const float*)d_in[3];   // [128,1]
    float* out = (float*)d_out;                // [8,2048,64]

    char* ws = (char*)d_ws;
    float*    Wh  = (float*)ws;                                   // 4 MB
    _Float16* WhT = (_Float16*)(ws + (size_t)16384 * 64 * 4);     // 2 MB
    float*    Wh1 = (float*)(ws + (size_t)6 * 1024 * 1024);       // 64 KB
    float*    Wh2 = Wh1 + 16384;                                  // 64 KB

    wh_kernel  <<< 256, 128, 0, stream>>>(X, Ws, Wh, WhT);
    wh12_kernel<<<  64, 256, 0, stream>>>(Wh, av, Wh1, Wh2);
    // one block per 16-row query tile; 4 waves split the 2048-wide j dimension
    gat_kernel <<<1024, 128, 0, stream>>>(A, Wh1, Wh2, WhT, out);
}
